// ExchangeBlock_44478681317459
// MI455X (gfx1250) — matrix operand order repaired and checked by reading, not compiled, not run on hardware
//
#include <hip/hip_runtime.h>
#include <hip/hip_bf16.h>
#include <math.h>

// ---------- constants of the problem ----------
#define NN     8192
#define EE     32768
#define BB     8
#define L0_    64
#define L1_    32
#define L2_    16
#define DIM    240          // L0 + 3*L1 + 5*L2
#define SS     256
#define NB_    128
#define KTP    5376         // 64*64 + 32*32 + 16*16
#define KTP_T  168          // KTP/32
#define KIN    288          // 257 padded to 32
#define CUT    7.0f
#define PI_F   3.14159265358979323846f

typedef __attribute__((ext_vector_type(16))) __bf16  v16bf;
typedef __attribute__((ext_vector_type(8)))  float   v8f;
typedef __attribute__((ext_vector_type(4)))  unsigned u32x4;

union AFrag {
    v16bf  v;
    unsigned u[8];
    u32x4  q[2];
};

// WMMA bf16 16x16x32, f32 accumulate
__device__ __forceinline__ v8f wmma_bf16(v16bf a, v16bf b, v8f c) {
    return __builtin_amdgcn_wmma_f32_16x16x32_bf16(
        /*neg_a=*/false, a, /*neg_b=*/false, b,
        /*c_mod=*/(short)0, c, /*reuse_a=*/false, /*reuse_b=*/false);
}

// B fragment: packed layout, lane block = 16 bf16 = 32B -> two b128 loads
__device__ __forceinline__ v16bf load_bfrag(const __bf16* p) {
    AFrag f;
    f.q[0] = *(const u32x4*)(p);
    f.q[1] = *(const u32x4*)(p + 8);
    return f.v;
}

// A fragment from an LDS-resident bf16 row (row-major, this lane's M row).
// elem 2i+j holds K = ktile*32 + half*8 + (i<4 ? 2i+j : 16+2(i-4)+j)
__device__ __forceinline__ v16bf afrag_from_lds(const __bf16* row, int kt, int half) {
    AFrag f;
    int kbase = kt * 32 + half * 8;
#pragma unroll
    for (int i = 0; i < 8; ++i) {
        int ke = kbase + ((i < 4) ? (2 * i) : (8 + 2 * i));   // even K index
        f.u[i] = *(const unsigned*)(&row[ke]);                // 2 bf16 at once
    }
    return f.v;
}

__device__ __forceinline__ float silu_f(float x) {
    return x / (1.0f + __expf(-x));
}

// gfx1250 async global->LDS copy (16B per lane), tracked by ASYNCcnt.
// lds_off: LDS byte address (low 32 bits of generic shared pointer);
// gptr: 64-bit global address in a VGPR pair (GV addressing mode).
__device__ __forceinline__ void async_g2lds_b128(unsigned lds_off, const void* gptr) {
    asm volatile("global_load_async_to_lds_b128 %0, %1, off"
                 :: "v"(lds_off), "v"(gptr) : "memory");
}
__device__ __forceinline__ void wait_asynccnt0() {
    asm volatile("s_wait_asynccnt 0x0" ::: "memory");
}

// ---------------------------------------------------------------------------
// K0a: pack plain [K,N] f32 weight into WMMA-B fragment order (bf16), pad K.
// packed[((nt*Kt + kt)*32 + lane)*16 + e] = W[k][n]
// ---------------------------------------------------------------------------
__global__ __launch_bounds__(256) void pack_b_kernel(const float* __restrict__ W,
                                                     __bf16* __restrict__ dst,
                                                     int K, int Kp, int N) {
    int idx = blockIdx.x * 256 + threadIdx.x;
    if (idx >= Kp * N) return;
    int k = idx / N, n = idx % N;
    float val = (k < K) ? W[(size_t)k * N + n] : 0.0f;
    int Kt = Kp / 32;
    int kt = k / 32, kin = k % 32;
    int nt = n / 16, nin = n % 16;
    int lane = ((kin / 8) % 2) * 16 + nin;
    int e    = (kin % 8) + (kin / 16) * 8;
    dst[(((size_t)nt * Kt + kt) * 32 + lane) * 16 + e] = (__bf16)val;
}

// ---------------------------------------------------------------------------
// K0b: build symmetrized + scaled concatenated TP weight Wcat[5376,256] packed
// ---------------------------------------------------------------------------
__global__ __launch_bounds__(256) void pack_wcat_kernel(const float* __restrict__ W0,
                                                        const float* __restrict__ W1,
                                                        const float* __restrict__ W2,
                                                        __bf16* __restrict__ dst) {
    int idx = blockIdx.x * 256 + threadIdx.x;
    if (idx >= KTP * SS) return;
    int k = idx / SS, n = idx % SS;
    const float alpha = rsqrtf((float)(L0_ * L0_ + L1_ * L1_ + L2_ * L2_));
    float val;
    if (k < 4096) {                       // l = 0 block
        int u = k >> 6, v = k & 63;
        val = 0.5f * (W0[((size_t)u * 64 + v) * SS + n] +
                      W0[((size_t)v * 64 + u) * SS + n]) * alpha;
    } else if (k < 5120) {                // l = 1 block
        int kk = k - 4096;
        int u = kk >> 5, v = kk & 31;
        val = 0.5f * (W1[((size_t)u * 32 + v) * SS + n] +
                      W1[((size_t)v * 32 + u) * SS + n]) * (alpha * rsqrtf(3.0f));
    } else {                              // l = 2 block
        int kk = k - 5120;
        int u = kk >> 4, v = kk & 15;
        val = 0.5f * (W2[((size_t)u * 16 + v) * SS + n] +
                      W2[((size_t)v * 16 + u) * SS + n]) * (alpha * rsqrtf(5.0f));
    }
    int kt = k / 32, kin = k % 32;
    int nt = n / 16, nin = n % 16;
    int lane = ((kin / 8) % 2) * 16 + nin;
    int e    = (kin % 8) + (kin / 16) * 8;
    dst[(((size_t)nt * KTP_T + kt) * 32 + lane) * 16 + e] = (__bf16)val;
}

// ---------------------------------------------------------------------------
// K0c: nodes f32 -> bf16
// ---------------------------------------------------------------------------
__global__ __launch_bounds__(256) void nodes_bf_kernel(const float* __restrict__ nodes,
                                                       __bf16* __restrict__ dst, int total) {
    int i = blockIdx.x * 256 + threadIdx.x;
    if (i < total) dst[i] = (__bf16)nodes[i];
}

// ---------------------------------------------------------------------------
// K1: edge geometry -> dist, exp(-dist); init out = b_out
// ---------------------------------------------------------------------------
__global__ __launch_bounds__(256) void edge_geom_kernel(const float* __restrict__ pos,
                                                        const float* __restrict__ cell,
                                                        const float* __restrict__ shift,
                                                        const int* __restrict__ eidx,
                                                        const int* __restrict__ batch,
                                                        const float* __restrict__ b_out,
                                                        float* __restrict__ dist,
                                                        float* __restrict__ expd,
                                                        float* __restrict__ out) {
    int e = blockIdx.x * 256 + threadIdx.x;
    if (e >= EE) return;
    int s = eidx[e], d = eidx[EE + e];
    const float* C = cell + (size_t)batch[s] * 9;
    float s0 = shift[e * 3 + 0], s1 = shift[e * 3 + 1], s2 = shift[e * 3 + 2];
    float t0 = s0 * C[0] + s1 * C[3] + s2 * C[6];
    float t1 = s0 * C[1] + s1 * C[4] + s2 * C[7];
    float t2 = s0 * C[2] + s1 * C[5] + s2 * C[8];
    float r0 = pos[d * 3 + 0] - pos[s * 3 + 0] + t0;
    float r1 = pos[d * 3 + 1] - pos[s * 3 + 1] + t1;
    float r2 = pos[d * 3 + 2] - pos[s * 3 + 2] + t2;
    float dd = sqrtf(r0 * r0 + r1 * r1 + r2 * r2);
    dist[e] = dd;
    expd[e] = __expf(-dd);
    out[e]  = b_out[0];
}

// ---------------------------------------------------------------------------
// K2: TP-GEMM  mixed[E,256] = Gram(A) x Wcat  + fused LayerNorm -> bf16 ws
//     block = 16 edges, 8 waves, each wave handles two 16-col tiles.
//     K loop split per irrep region (l=0/1/2) - no per-step branching.
// ---------------------------------------------------------------------------
__global__ __launch_bounds__(256) void tp_gemm_kernel(const __bf16* __restrict__ nodes_bf,
                                                      const int* __restrict__ eidx,
                                                      const __bf16* __restrict__ wcat,
                                                      const float* __restrict__ gamma,
                                                      const float* __restrict__ beta,
                                                      __bf16* __restrict__ mixednorm) {
    __shared__ __align__(16) float xs[16][DIM];
    __shared__ __align__(16) float xd[16][DIM];
    __shared__ __align__(16) float mix[16][SS];
    __shared__ float red1[16][16];
    __shared__ float red2[16][16];

    const int e0  = blockIdx.x * 16;
    const int tid = threadIdx.x;

    for (int i = tid; i < 16 * DIM; i += 256) {
        int m = i / DIM, c = i % DIM;
        int es = eidx[e0 + m], ed = eidx[EE + e0 + m];
        xs[m][c] = (float)nodes_bf[(size_t)es * DIM + c];
        xd[m][c] = (float)nodes_bf[(size_t)ed * DIM + c];
    }
    __syncthreads();

    const int wave = tid / 32, lane = tid % 32;
    const int m = lane % 16, half = lane / 16, nn = lane % 16;
    const int hoff = half * 8;
    const float* rs = xs[m];
    const float* rd = xd[m];

    v8f acc0 = {}, acc1 = {};
    const int nt0 = wave * 2;
    const __bf16* bbase0 = wcat + ((size_t)(nt0 + 0) * KTP_T * 32 + lane) * 16;
    const __bf16* bbase1 = wcat + ((size_t)(nt0 + 1) * KTP_T * 32 + lane) * 16;

    // ---- l = 0 region: kt 0..127, A[k]=rs[k>>6]*rd[k&63] ----
    for (int kt = 0; kt < 128; ++kt) {
        AFrag a;
        int kbase = kt * 32 + hoff;
#pragma unroll
        for (int i = 0; i < 8; ++i) {
            int ke = kbase + ((i < 4) ? (2 * i) : (8 + 2 * i));
            float p0 = rs[ke >> 6] * rd[ke & 63];
            float p1 = rs[(ke + 1) >> 6] * rd[(ke + 1) & 63];
            a.v[2 * i + 0] = (__bf16)p0;
            a.v[2 * i + 1] = (__bf16)p1;
        }
        v16bf b0 = load_bfrag(bbase0 + (size_t)kt * 512);
        v16bf b1 = load_bfrag(bbase1 + (size_t)kt * 512);
        __builtin_prefetch(bbase0 + (size_t)(kt + 4) * 512, 0, 1);
        __builtin_prefetch(bbase1 + (size_t)(kt + 4) * 512, 0, 1);
        acc0 = wmma_bf16(a.v, b0, acc0);
        acc1 = wmma_bf16(a.v, b1, acc1);
    }
    // ---- l = 1 region: kt 128..159, dot3 over block at col 64 ----
    for (int kt = 128; kt < 160; ++kt) {
        AFrag a;
        int kbase = kt * 32 + hoff - 4096;
#pragma unroll
        for (int i = 0; i < 8; ++i) {
            int kk = kbase + ((i < 4) ? (2 * i) : (8 + 2 * i));
            const float* su = rs + 64 + 3 * (kk >> 5);
            const float* dv = rd + 64 + 3 * (kk & 31);
            float p0 = su[0] * dv[0] + su[1] * dv[1] + su[2] * dv[2];
            int k1 = kk + 1;
            su = rs + 64 + 3 * (k1 >> 5);
            dv = rd + 64 + 3 * (k1 & 31);
            float p1 = su[0] * dv[0] + su[1] * dv[1] + su[2] * dv[2];
            a.v[2 * i + 0] = (__bf16)p0;
            a.v[2 * i + 1] = (__bf16)p1;
        }
        v16bf b0 = load_bfrag(bbase0 + (size_t)kt * 512);
        v16bf b1 = load_bfrag(bbase1 + (size_t)kt * 512);
        acc0 = wmma_bf16(a.v, b0, acc0);
        acc1 = wmma_bf16(a.v, b1, acc1);
    }
    // ---- l = 2 region: kt 160..167, dot5 over block at col 160 ----
    for (int kt = 160; kt < 168; ++kt) {
        AFrag a;
        int kbase = kt * 32 + hoff - 5120;
#pragma unroll
        for (int i = 0; i < 8; ++i) {
            int kk = kbase + ((i < 4) ? (2 * i) : (8 + 2 * i));
            const float* su = rs + 160 + 5 * (kk >> 4);
            const float* dv = rd + 160 + 5 * (kk & 15);
            float p0 = su[0]*dv[0] + su[1]*dv[1] + su[2]*dv[2] + su[3]*dv[3] + su[4]*dv[4];
            int k1 = kk + 1;
            su = rs + 160 + 5 * (k1 >> 4);
            dv = rd + 160 + 5 * (k1 & 15);
            float p1 = su[0]*dv[0] + su[1]*dv[1] + su[2]*dv[2] + su[3]*dv[3] + su[4]*dv[4];
            a.v[2 * i + 0] = (__bf16)p0;
            a.v[2 * i + 1] = (__bf16)p1;
        }
        v16bf b0 = load_bfrag(bbase0 + (size_t)kt * 512);
        v16bf b1 = load_bfrag(bbase1 + (size_t)kt * 512);
        acc0 = wmma_bf16(a.v, b0, acc0);
        acc1 = wmma_bf16(a.v, b1, acc1);
    }

#pragma unroll
    for (int r = 0; r < 8; ++r) {
        mix[r + hoff][(nt0 + 0) * 16 + nn] = acc0[r];
        mix[r + hoff][(nt0 + 1) * 16 + nn] = acc1[r];
    }
    __syncthreads();

    // fused LayerNorm over S=256, parallel across all 256 threads
    const int eL = tid >> 4;      // edge 0..15
    const int sg = tid & 15;      // 16-element segment
    {
        float s = 0.0f, s2 = 0.0f;
#pragma unroll
        for (int j = sg * 16; j < sg * 16 + 16; ++j) {
            float x = mix[eL][j];
            s += x; s2 += x * x;
        }
        red1[eL][sg] = s;
        red2[eL][sg] = s2;
    }
    __syncthreads();
    if (tid < 16) {
        float a = 0.0f, b = 0.0f;
#pragma unroll
        for (int j = 0; j < 16; ++j) { a += red1[tid][j]; b += red2[tid][j]; }
        float mu  = a * (1.0f / SS);
        float var = b * (1.0f / SS) - mu * mu;
        red1[tid][0] = mu;
        red2[tid][0] = rsqrtf(var + 1e-5f);
    }
    __syncthreads();
    {
        float mu  = red1[eL][0];
        float inv = red2[eL][0];
        size_t base = (size_t)(e0 + eL) * SS;
        for (int j = sg * 16; j < sg * 16 + 16; ++j) {
            float y = (mix[eL][j] - mu) * inv * gamma[j] + beta[j];
            mixednorm[base + j] = (__bf16)y;
        }
    }
}

// ---------------------------------------------------------------------------
// K3: fused radial filter MLP per 16-edge tile:
//     rbf[16,128] -> silu(rbf@Wf1+b) -> @Wf2+b -> * mixednorm -> mlp_in[E,288]
// ---------------------------------------------------------------------------
__global__ __launch_bounds__(256) void filter_mlp_kernel(const float* __restrict__ dist,
                                                         const float* __restrict__ expd,
                                                         const __bf16* __restrict__ wf1p,
                                                         const float* __restrict__ bf1,
                                                         const __bf16* __restrict__ wf2p,
                                                         const float* __restrict__ bf2,
                                                         const __bf16* __restrict__ mixednorm,
                                                         __bf16* __restrict__ mlp_in) {
    __shared__ __align__(16) __bf16 rbf[16][NB_];
    __shared__ __align__(16) __bf16 hbuf[16][512];
    __shared__ float dsh[16];

    const int e0 = blockIdx.x * 16;
    const int tid = threadIdx.x;
    if (tid < 16) dsh[tid] = dist[e0 + tid];
    __syncthreads();

    const float cnorm = sqrtf(2.0f / CUT);
    for (int i = tid; i < 16 * NB_; i += 256) {
        int m = i / NB_, n = i % NB_;
        float d = dsh[m];
        rbf[m][n] = (__bf16)(cnorm * __sinf((float)(n + 1) * PI_F * d / CUT) / d);
    }
    __syncthreads();

    const int wave = tid / 32, lane = tid % 32;
    const int m = lane % 16, half = lane / 16, nn = lane % 16;

    // GEMM1: [16,128] x [128,512]
    for (int t = 0; t < 4; ++t) {
        int nt = wave * 4 + t;
        v8f acc = {};
#pragma unroll
        for (int kt = 0; kt < 4; ++kt) {
            v16bf a = afrag_from_lds(&rbf[m][0], kt, half);
            v16bf b = load_bfrag(wf1p + (((size_t)nt * 4 + kt) * 32 + lane) * 16);
            acc = wmma_bf16(a, b, acc);
        }
#pragma unroll
        for (int r = 0; r < 8; ++r) {
            int row = r + half * 8, col = nt * 16 + nn;
            hbuf[row][col] = (__bf16)silu_f(acc[r] + bf1[col]);
        }
    }
    __syncthreads();

    // GEMM2: [16,512] x [512,256] + epilogue: * mixednorm
    for (int t = 0; t < 2; ++t) {
        int nt = wave * 2 + t;
        v8f acc = {};
        for (int kt = 0; kt < 16; ++kt) {
            v16bf a = afrag_from_lds(&hbuf[m][0], kt, half);
            v16bf b = load_bfrag(wf2p + (((size_t)nt * 16 + kt) * 32 + lane) * 16);
            acc = wmma_bf16(a, b, acc);
        }
#pragma unroll
        for (int r = 0; r < 8; ++r) {
            int row = r + half * 8, col = nt * 16 + nn;
            float df = acc[r] + bf2[col];
            float mn = (float)mixednorm[(size_t)(e0 + row) * SS + col];
            mlp_in[(size_t)(e0 + row) * KIN + col] = (__bf16)(mn * df);
        }
    }
    // columns 256 (=exp(-dist)) and 257..287 (=0)
    for (int i = tid; i < 16 * 32; i += 256) {
        int mm = i / 32, c = i % 32;
        float v = (c == 0) ? expd[e0 + mm] : 0.0f;
        mlp_in[(size_t)(e0 + mm) * KIN + 256 + c] = (__bf16)v;
    }
}

// ---------------------------------------------------------------------------
// K5: h1 = silu(mlp_in[E,288] @ W_in + b_in)  -> bf16 [E,1024]
//     A tile staged via gfx1250 async global->LDS DMA.
// ---------------------------------------------------------------------------
__global__ __launch_bounds__(256) void edge_mlp1_kernel(const __bf16* __restrict__ mlp_in,
                                                        const __bf16* __restrict__ winp,
                                                        const float* __restrict__ b_in,
                                                        __bf16* __restrict__ h1) {
    __shared__ __align__(16) __bf16 ain[16][KIN];

    const int e0 = blockIdx.x * 16;
    const int tid = threadIdx.x;

    {   // async stage 16*288*2 = 9216 B = 576 x b128
        const char* src = (const char*)(mlp_in + (size_t)e0 * KIN);
        unsigned lds0 = (unsigned)(uintptr_t)(&ain[0][0]);
        for (int i = tid; i < 576; i += 256)
            async_g2lds_b128(lds0 + i * 16, src + (size_t)i * 16);
        wait_asynccnt0();
    }
    __syncthreads();

    const int wave = tid / 32, lane = tid % 32;
    const int m = lane % 16, half = lane / 16, nn = lane % 16;
    const int Kt = KIN / 32;   // 9

    for (int t = 0; t < 8; ++t) {
        int nt = wave * 8 + t;
        v8f acc = {};
        for (int kt = 0; kt < Kt; ++kt) {
            v16bf a = afrag_from_lds(&ain[m][0], kt, half);
            v16bf b = load_bfrag(winp + (((size_t)nt * Kt + kt) * 32 + lane) * 16);
            acc = wmma_bf16(a, b, acc);
        }
#pragma unroll
        for (int r = 0; r < 8; ++r) {
            int row = r + half * 8, col = nt * 16 + nn;
            h1[(size_t)(e0 + row) * 1024 + col] = (__bf16)silu_f(acc[r] + b_in[col]);
        }
    }
}

// ---------------------------------------------------------------------------
// K6: h2 = silu(h1 @ W_res + b_res); out[e] += (h1+h2) . W_out
//     grid.y splits the 1024 output cols into 8 groups of 128.
//     h1 tile (32 KB) staged via gfx1250 async global->LDS DMA.
// ---------------------------------------------------------------------------
__global__ __launch_bounds__(256) void edge_mlp2_kernel(const __bf16* __restrict__ h1,
                                                        const __bf16* __restrict__ wresp,
                                                        const float* __restrict__ b_res,
                                                        const float* __restrict__ w_out,
                                                        float* __restrict__ out) {
    __shared__ __align__(16) __bf16 ah[16][1024];
    __shared__ float red[16];

    const int e0 = blockIdx.x * 16;
    const int tid = threadIdx.x;

    {   // async stage 16*1024*2 = 32768 B = 2048 x b128
        const char* src = (const char*)(h1 + (size_t)e0 * 1024);
        unsigned lds0 = (unsigned)(uintptr_t)(&ah[0][0]);
#pragma unroll
        for (int rep = 0; rep < 8; ++rep) {
            int i = rep * 256 + tid;
            async_g2lds_b128(lds0 + i * 16, src + (size_t)i * 16);
        }
        wait_asynccnt0();
    }
    if (tid < 16) red[tid] = 0.0f;
    __syncthreads();

    const int wave = tid / 32, lane = tid % 32;
    const int m = lane % 16, half = lane / 16, nn = lane % 16;
    const int nt = blockIdx.y * 8 + wave;

    v8f acc = {};
    for (int kt = 0; kt < 32; ++kt) {
        v16bf a = afrag_from_lds(&ah[m][0], kt, half);
        v16bf b = load_bfrag(wresp + (((size_t)nt * 32 + kt) * 32 + lane) * 16);
        acc = wmma_bf16(a, b, acc);
    }
#pragma unroll
    for (int r = 0; r < 8; ++r) {
        int row = r + half * 8, col = nt * 16 + nn;
        float h2 = silu_f(acc[r] + b_res[col]);
        atomicAdd(&red[row], h2 * w_out[col]);
    }
    __syncthreads();

    if (tid < 16) atomicAdd(out + e0 + tid, red[tid]);

    // h1 . W_out counted exactly once (block y == 0 has the full h1 tile in LDS)
    if (blockIdx.y == 0 && tid < 16) {
        float s = 0.0f;
        const __bf16* hr = &ah[tid][0];
        for (int j = 0; j < 1024; ++j) s += (float)hr[j] * w_out[j];
        atomicAdd(out + e0 + tid, s);
    }
}

// ---------------------------------------------------------------------------
// host
// ---------------------------------------------------------------------------
static inline size_t align256(size_t x) { return (x + 255) & ~(size_t)255; }

extern "C" void kernel_launch(void* const* d_in, const int* in_sizes, int n_in,
                              void* d_out, int out_size, void* d_ws, size_t ws_size,
                              hipStream_t stream) {
    (void)in_sizes; (void)n_in; (void)out_size; (void)ws_size;

    const float* nodes = (const float*)d_in[0];
    const float* pos   = (const float*)d_in[1];
    const float* cell  = (const float*)d_in[2];
    const float* shift = (const float*)d_in[3];
    const int*   eidx  = (const int*)d_in[4];
    const int*   batch = (const int*)d_in[5];
    const float* W0    = (const float*)d_in[6];
    const float* W1    = (const float*)d_in[7];
    const float* W2    = (const float*)d_in[8];
    const float* gamma = (const float*)d_in[9];
    const float* beta  = (const float*)d_in[10];
    const float* Wf1   = (const float*)d_in[11];
    const float* bf1   = (const float*)d_in[12];
    const float* Wf2   = (const float*)d_in[13];
    const float* bf2   = (const float*)d_in[14];
    const float* W_in  = (const float*)d_in[15];
    const float* b_in  = (const float*)d_in[16];
    const float* W_res = (const float*)d_in[17];
    const float* b_res = (const float*)d_in[18];
    const float* W_out = (const float*)d_in[19];
    const float* b_out = (const float*)d_in[20];
    float* out = (float*)d_out;

    char* ws = (char*)d_ws;
    size_t off = 0;
    __bf16* wcat_p = (__bf16*)(ws + off); off = align256(off + (size_t)KTP * SS * 2);
    __bf16* wf1_p  = (__bf16*)(ws + off); off = align256(off + (size_t)128 * 512 * 2);
    __bf16* wf2_p  = (__bf16*)(ws + off); off = align256(off + (size_t)512 * 256 * 2);
    __bf16* win_p  = (__bf16*)(ws + off); off = align256(off + (size_t)KIN * 1024 * 2);
    __bf16* wres_p = (__bf16*)(ws + off); off = align256(off + (size_t)1024 * 1024 * 2);
    __bf16* nodesb = (__bf16*)(ws + off); off = align256(off + (size_t)NN * DIM * 2);
    float*  dist   = (float*)(ws + off);  off = align256(off + (size_t)EE * 4);
    float*  expd   = (float*)(ws + off);  off = align256(off + (size_t)EE * 4);
    __bf16* mixn   = (__bf16*)(ws + off); off = align256(off + (size_t)EE * SS * 2);
    __bf16* mlpin  = (__bf16*)(ws + off); off = align256(off + (size_t)EE * KIN * 2);
    __bf16* h1     = (__bf16*)(ws + off); off = align256(off + (size_t)EE * 1024 * 2);

    // weight prep
    pack_wcat_kernel<<<(KTP * SS) / 256, 256, 0, stream>>>(W0, W1, W2, wcat_p);
    pack_b_kernel<<<(128 * 512) / 256, 256, 0, stream>>>(Wf1, wf1_p, 128, 128, 512);
    pack_b_kernel<<<(512 * 256) / 256, 256, 0, stream>>>(Wf2, wf2_p, 512, 512, 256);
    pack_b_kernel<<<(KIN * 1024) / 256, 256, 0, stream>>>(W_in, win_p, 257, KIN, 1024);
    pack_b_kernel<<<(1024 * 1024) / 256, 256, 0, stream>>>(W_res, wres_p, 1024, 1024, 1024);
    nodes_bf_kernel<<<(NN * DIM + 255) / 256, 256, 0, stream>>>(nodes, nodesb, NN * DIM);

    // pipeline
    edge_geom_kernel<<<EE / 256, 256, 0, stream>>>(pos, cell, shift, eidx, batch,
                                                   b_out, dist, expd, out);
    tp_gemm_kernel<<<EE / 16, 256, 0, stream>>>(nodesb, eidx, wcat_p, gamma, beta, mixn);
    filter_mlp_kernel<<<EE / 16, 256, 0, stream>>>(dist, expd, wf1_p, bf1, wf2_p, bf2,
                                                   mixn, mlpin);
    edge_mlp1_kernel<<<EE / 16, 256, 0, stream>>>(mlpin, win_p, b_in, h1);
    edge_mlp2_kernel<<<dim3(EE / 16, 8), 256, 0, stream>>>(h1, wres_p, b_res, W_out, out);
}
